// AttentionPolicyHead_63118839382360
// MI455X (gfx1250) — compile-verified, matrix operand order checked by reading, not run on hardware
//
#include <hip/hip_runtime.h>
#include <hip/hip_bf16.h>

// ---------------------------------------------------------------------------
// CDNA5 / gfx1250 fused attention-policy head.
// Matrix math: v_wmma_f32_16x16x32_bf16 (bf16 in, f32 accumulate).
// Weight/policy-map tiles staged LDS-side with GLOBAL_LOAD_ASYNC_TO_LDS_B128,
// triple-buffered, synchronized with s_wait_asynccnt + workgroup barriers.
// ---------------------------------------------------------------------------

typedef __bf16 bf16_t;
typedef __bf16 v16bf __attribute__((ext_vector_type(16)));
typedef __bf16 v8bf  __attribute__((ext_vector_type(8)));
typedef float  v8f   __attribute__((ext_vector_type(8)));

#define HID   1024
#define NSQ   64
#define BATCH 2048
#define KPOL  4288   // 64*64 + 8*24
#define NPOL  1858

#define HS_STRIDE 1032   // 64x1024 bf16 hs tile in LDS, padded row stride
#define QC_STRIDE 136    // 64x128 bf16 Q/K chunk in LDS, padded row stride
#define WST      72      // staged 128x64 slice row stride (72*2B=144B -> no 16-row bank clash)
#define SLICE    (128 * WST)

__device__ __forceinline__ v8f wmma_bf16(v16bf a, v16bf b, v8f c) {
  return __builtin_amdgcn_wmma_f32_16x16x32_bf16(false, a, false, b,
                                                 (short)0, c, false, false);
}

__device__ __forceinline__ v8f vzero8() {
  v8f z;
#pragma unroll
  for (int i = 0; i < 8; ++i) z[i] = 0.0f;
  return z;
}

// Async 16B global->LDS copy (ASYNCcnt-tracked). lds_off = low 32 bits of the
// generic pointer to __shared__ (the LDS byte offset).
__device__ __forceinline__ void async_ld_b128(unsigned lds_off, const void* g) {
  asm volatile("global_load_async_to_lds_b128 %0, %1, off"
               :: "v"(lds_off), "v"(g) : "memory");
}
__device__ __forceinline__ unsigned lds_addr(const void* p) {
  return (unsigned)(unsigned long long)p;
}
__device__ __forceinline__ void wait_async4() {  // keep next slice in flight
  asm volatile("s_wait_asynccnt 0x4" ::: "memory");
}
__device__ __forceinline__ void wait_async0() {  // drain
  asm volatile("s_wait_asynccnt 0x0" ::: "memory");
}

// A/B fragment gather from a row-major bf16 matrix (LDS or global).
// ISA 16-bit A layout (16x32): lane L -> row = L%16, kbase = (L/16)*8;
// elems 0..7 = K kbase..kbase+7, elems 8..15 = K kbase+16..kbase+23.
// B operand loaded from B^T rows with the symmetric layout (N = L%16).
__device__ __forceinline__ v16bf ldfrag(const bf16_t* p0, int stride) {
  const int lane = threadIdx.x & 31;
  const bf16_t* p = p0 + (size_t)(lane & 15) * stride + ((lane >> 4) << 3);
  v8bf lo = *(const v8bf*)p;
  v8bf hi = *(const v8bf*)(p + 16);
  return __builtin_shufflevector(lo, hi, 0, 1, 2, 3, 4, 5, 6, 7,
                                 8, 9, 10, 11, 12, 13, 14, 15);
}

// C/D layout: VGPR r, lanes 0-15: M=r, N=lane; lanes 16-31: M=8+r, N=lane-16.
__device__ __forceinline__ void stfrag_bf16(bf16_t* p0, int stride, v8f acc) {
  const int lane = threadIdx.x & 31;
  bf16_t* p = p0 + (lane & 15) + (size_t)((lane >> 4) << 3) * stride;
#pragma unroll
  for (int r = 0; r < 8; ++r) p[(size_t)r * stride] = (bf16_t)acc[r];
}

// ---------------------------------------------------------------------------
// Prep kernels
// ---------------------------------------------------------------------------
__global__ void conv_w_kernel(const float* __restrict__ Wq,
                              const float* __restrict__ Wk,
                              bf16_t* __restrict__ WqB,
                              bf16_t* __restrict__ WkB, int n) {
  int i = blockIdx.x * blockDim.x + threadIdx.x;
  if (i < n) {
    WqB[i] = (bf16_t)Wq[i];
    WkB[i] = (bf16_t)Wk[i];
  }
}

// policy_map (4288 x 1858 f32) -> pmT (1858 x 4288 bf16), LDS-tiled transpose
__global__ void transpose_pm_kernel(const float* __restrict__ pm,
                                    bf16_t* __restrict__ pmT) {
  __shared__ float tile[32][33];
  const int nBase = blockIdx.x * 32;
  const int kBase = blockIdx.y * 32;
  const int tx = threadIdx.x, ty = threadIdx.y;
#pragma unroll
  for (int j = 0; j < 32; j += 8) {
    int kk = kBase + ty + j, nn = nBase + tx;
    tile[ty + j][tx] =
        (kk < KPOL && nn < NPOL) ? pm[(size_t)kk * NPOL + nn] : 0.0f;
  }
  __syncthreads();
#pragma unroll
  for (int j = 0; j < 32; j += 8) {
    int nn = nBase + ty + j, kk = kBase + tx;
    if (nn < NPOL && kk < KPOL)
      pmT[(size_t)nn * KPOL + kk] = (bf16_t)tile[tx][ty + j];
  }
}

// Wfused = Wpo @ Wpk  (4 x 1024), so promo_offsets = hs[:,48:56] @ Wfused^T
__global__ void wfused_kernel(const float* __restrict__ Wpo,
                              const float* __restrict__ Wpk,
                              float* __restrict__ Wf) {
  int idx = blockIdx.x * blockDim.x + threadIdx.x;  // 0..4095
  int i = idx >> 10, d = idx & (HID - 1);
  float s = 0.0f;
  for (int c = 0; c < HID; ++c) s += Wpo[i * HID + c] * Wpk[c * HID + d];
  Wf[idx] = s;
}

// ---------------------------------------------------------------------------
// Fused per-board kernel: one block per batch element, 512 threads = 16 waves.
// ~279 KB LDS (gfx1250: 320 KB/workgroup) -> 1 workgroup per WGP.
// ---------------------------------------------------------------------------
__global__ __launch_bounds__(512) void fused_attention_kernel(
    const float* __restrict__ hs, const bf16_t* __restrict__ WqB,
    const bf16_t* __restrict__ WkB, const float* __restrict__ Wfused,
    float* __restrict__ pa_out, bf16_t* __restrict__ combined) {
  __shared__ __align__(16) bf16_t hsS[NSQ * HS_STRIDE];                   // 132 KB
  __shared__ __align__(16) unsigned char qkRaw[2 * NSQ * QC_STRIDE * 2];  //  34 KB
  __shared__ __align__(16) bf16_t wstage[3 * 2 * SLICE];                  // 108 KB
  __shared__ float partialS[24 * 16];

  bf16_t* Qc = (bf16_t*)qkRaw;
  bf16_t* Kc = Qc + NSQ * QC_STRIDE;
  float* SS = (float*)qkRaw;  // 64x64 f32 scores overlay (reused after GEMMs)

  const int b = blockIdx.x;
  const int tid = threadIdx.x;
  const int w = tid >> 5;   // wave 0..15
  const int lane = tid & 31;
  const int mi = w >> 2;    // M-tile
  const int ni = w & 3;     // N-tile

  // Cooperative async-issue of one 128-row x 64-K slice of Wq and Wk.
  // Each thread moves 2x16B per matrix; fully coalesced global_load_b128s.
  const int srow = tid >> 3;        // 0..63 (+64 on second half)
  const int scol = (tid & 7) * 8;   // 0..56
  auto issue_wslice = [&](int buf, int rowbase, int k) {
    bf16_t* dq = wstage + (size_t)(buf * 2 + 0) * SLICE;
    bf16_t* dk = wstage + (size_t)(buf * 2 + 1) * SLICE;
#pragma unroll
    for (int h = 0; h < 2; ++h) {
      int row = srow + h * 64;
      const size_t goff = (size_t)(rowbase + row) * HID + k + scol;
      unsigned l = (unsigned)(row * WST + scol) * 2u;
      async_ld_b128(lds_addr(dq) + l, WqB + goff);
      async_ld_b128(lds_addr(dk) + l, WkB + goff);
    }
  };

  // Phase 0: hs_b (64x1024 f32) -> bf16 in LDS (HBM read exactly once)
  const float4* hs4 = (const float4*)(hs + (size_t)b * NSQ * HID);
  for (int e = tid; e < NSQ * HID / 4; e += 512) {
    float4 v = hs4[e];
    int idx = e * 4;
    bf16_t* d = &hsS[(idx >> 10) * HS_STRIDE + (idx & (HID - 1))];
    d[0] = (bf16_t)v.x; d[1] = (bf16_t)v.y;
    d[2] = (bf16_t)v.z; d[3] = (bf16_t)v.w;
  }
  __syncthreads();

  // Persistent 16x16 score accumulator per wave (64x64 = 4x4 wave tiles)
  v8f Sacc = vzero8();

  // 8 chunks of 128 Q/K features; projections + score accumulation fused.
  for (int t = 0; t < 8; ++t) {
    const int rowbase = t * 128;
    issue_wslice(0, rowbase, 0);  // prologue slice
    v8f aQ0 = vzero8(), aQ1 = vzero8(), aK0 = vzero8(), aK1 = vzero8();
    for (int i = 0; i < 16; ++i) {  // 16 slices of K=64
      if (i + 1 < 16) {
        issue_wslice((i + 1) % 3, rowbase, (i + 1) * 64);
        wait_async4();   // this slice's 4 ops may stay in flight; prev landed
      } else {
        wait_async0();
      }
      __syncthreads();   // slice i visible to whole workgroup
      const bf16_t* wq = wstage + (size_t)((i % 3) * 2 + 0) * SLICE;
      const bf16_t* wk = wstage + (size_t)((i % 3) * 2 + 1) * SLICE;
#pragma unroll
      for (int kk = 0; kk < 64; kk += 32) {
        v16bf af  = ldfrag(&hsS[(mi * 16) * HS_STRIDE + i * 64 + kk], HS_STRIDE);
        v16bf bq0 = ldfrag(wq + (ni * 16) * WST + kk, WST);
        v16bf bq1 = ldfrag(wq + ((ni + 4) * 16) * WST + kk, WST);
        v16bf bk0 = ldfrag(wk + (ni * 16) * WST + kk, WST);
        v16bf bk1 = ldfrag(wk + ((ni + 4) * 16) * WST + kk, WST);
        aQ0 = wmma_bf16(af, bq0, aQ0);
        aQ1 = wmma_bf16(af, bq1, aQ1);
        aK0 = wmma_bf16(af, bk0, aK0);
        aK1 = wmma_bf16(af, bk1, aK1);
      }
    }
    stfrag_bf16(Qc + (mi * 16) * QC_STRIDE + ni * 16, QC_STRIDE, aQ0);
    stfrag_bf16(Qc + (mi * 16) * QC_STRIDE + (ni + 4) * 16, QC_STRIDE, aQ1);
    stfrag_bf16(Kc + (mi * 16) * QC_STRIDE + ni * 16, QC_STRIDE, aK0);
    stfrag_bf16(Kc + (mi * 16) * QC_STRIDE + (ni + 4) * 16, QC_STRIDE, aK1);
    __syncthreads();
    // S += Qc * Kc^T  (B operand = Kc rows, symmetric fragment load)
#pragma unroll
    for (int kk = 0; kk < 128; kk += 32) {
      v16bf aq = ldfrag(Qc + (mi * 16) * QC_STRIDE + kk, QC_STRIDE);
      v16bf bk = ldfrag(Kc + (ni * 16) * QC_STRIDE + kk, QC_STRIDE);
      Sacc = wmma_bf16(aq, bk, Sacc);
    }
    __syncthreads();
  }

  // policy_attention = raw_dot * (1/16 heads) * (1/sqrt(64)) = raw/128
  {
    float* q = SS + (size_t)(mi * 16 + ((lane >> 4) << 3)) * NSQ +
               ni * 16 + (lane & 15);
#pragma unroll
    for (int r = 0; r < 8; ++r) q[(size_t)r * NSQ] = Sacc[r] * (1.0f / 128.0f);
  }
  __syncthreads();

  // policy_attention (f32 out) and flat_att (= PA/8) into combined (bf16)
  float* paB = pa_out + (size_t)b * 4096;
  bf16_t* cmbB = combined + (size_t)b * KPOL;
  for (int idx = tid; idx < 4096; idx += 512) {
    float v = SS[idx];
    paB[idx] = v;
    cmbB[idx] = (bf16_t)(v * 0.125f);
  }

  // Promotion offsets: off[i][f] = hs[b,48+f,:] . Wfused[i,:]  (i<3, f<8)
  if (tid < 384) {
    int p = tid >> 4, s = tid & 15;  // p = i*8+f
    const float* wf = Wfused + (p >> 3) * HID;
    const bf16_t* hrow = &hsS[(48 + (p & 7)) * HS_STRIDE];
    float acc = 0.0f;
    for (int j = s; j < HID; j += 16) acc += (float)hrow[j] * wf[j];
    partialS[p * 16 + s] = acc;
  }
  __syncthreads();
  if (tid < 24) {
    int i = tid >> 3, f = tid & 7;
    float off = 0.0f;
#pragma unroll
    for (int s = 0; s < 16; ++s) off += partialS[tid * 16 + s];
    float so = off * 8.0f;  // * sqrt(dk)
#pragma unroll
    for (int p2 = 0; p2 < 8; ++p2) {
      float base = SS[(48 + p2) * NSQ + 56 + f];
      cmbB[4096 + p2 * 24 + f * 3 + i] = (bf16_t)(base + so);
    }
  }
}

// ---------------------------------------------------------------------------
// policy_logits = combined (2048x4288 bf16) @ policy_map  via pmT (1858x4288)
// 128x128 block tile; A and B^T slices async-staged to LDS, triple-buffered.
// K = 4288 = 67 slices of 64. ~111 KB LDS -> 2 workgroups/WGP.
// ---------------------------------------------------------------------------
__global__ __launch_bounds__(512) void policy_gemm_kernel(
    const bf16_t* __restrict__ A, const bf16_t* __restrict__ Bt,
    float* __restrict__ C) {
  __shared__ __align__(16) bf16_t astage[3 * SLICE];
  __shared__ __align__(16) bf16_t bstage[3 * SLICE];

  const int n0 = blockIdx.x * 128;
  const int m0 = blockIdx.y * 128;
  const int tid = threadIdx.x;
  const int w = tid >> 5;
  const int lane = tid & 31;
  const int mi = w >> 2, ni = w & 3;

  const int srow = tid >> 3;
  const int scol = (tid & 7) * 8;
  auto issue_slice = [&](int buf, int k) {
    bf16_t* da = astage + (size_t)buf * SLICE;
    bf16_t* db = bstage + (size_t)buf * SLICE;
#pragma unroll
    for (int h = 0; h < 2; ++h) {
      int row = srow + h * 64;
      unsigned l = (unsigned)(row * WST + scol) * 2u;
      int brow = n0 + row;
      if (brow >= NPOL) brow = NPOL - 1;  // duplicate row; masked on store
      async_ld_b128(lds_addr(da) + l, A + (size_t)(m0 + row) * KPOL + k + scol);
      async_ld_b128(lds_addr(db) + l, Bt + (size_t)brow * KPOL + k + scol);
    }
  };

  issue_slice(0, 0);
  v8f a00 = vzero8(), a01 = vzero8(), a10 = vzero8(), a11 = vzero8();
  for (int i = 0; i < 67; ++i) {
    if (i + 1 < 67) {
      issue_slice((i + 1) % 3, (i + 1) * 64);
      wait_async4();
    } else {
      wait_async0();
    }
    __syncthreads();
    const bf16_t* ast = astage + (size_t)(i % 3) * SLICE;
    const bf16_t* bst = bstage + (size_t)(i % 3) * SLICE;
#pragma unroll
    for (int kk = 0; kk < 64; kk += 32) {
      v16bf af0 = ldfrag(ast + (mi * 16) * WST + kk, WST);
      v16bf af1 = ldfrag(ast + ((mi + 4) * 16) * WST + kk, WST);
      v16bf bf0 = ldfrag(bst + (ni * 16) * WST + kk, WST);
      v16bf bf1 = ldfrag(bst + ((ni + 4) * 16) * WST + kk, WST);
      a00 = wmma_bf16(af0, bf0, a00);
      a01 = wmma_bf16(af0, bf1, a01);
      a10 = wmma_bf16(af1, bf0, a10);
      a11 = wmma_bf16(af1, bf1, a11);
    }
  }

  const int colBase0 = n0 + ni * 16 + (lane & 15);
  const int colBase1 = n0 + (ni + 4) * 16 + (lane & 15);
  const int rowOff = (lane >> 4) << 3;
#pragma unroll
  for (int r = 0; r < 8; ++r) {
    int row0 = m0 + mi * 16 + rowOff + r;
    int row1 = m0 + (mi + 4) * 16 + rowOff + r;
    if (colBase0 < NPOL) {
      C[(size_t)row0 * NPOL + colBase0] = a00[r];
      C[(size_t)row1 * NPOL + colBase0] = a10[r];
    }
    if (colBase1 < NPOL) {
      C[(size_t)row0 * NPOL + colBase1] = a01[r];
      C[(size_t)row1 * NPOL + colBase1] = a11[r];
    }
  }
}

// ---------------------------------------------------------------------------
extern "C" void kernel_launch(void* const* d_in, const int* in_sizes, int n_in,
                              void* d_out, int out_size, void* d_ws,
                              size_t ws_size, hipStream_t stream) {
  const float* hs  = (const float*)d_in[0];  // (2048, 64, 1024)
  const float* Wq  = (const float*)d_in[1];  // (1024, 1024)
  const float* Wk  = (const float*)d_in[2];  // (1024, 1024)
  const float* Wpk = (const float*)d_in[3];  // (1024, 1024)
  const float* Wpo = (const float*)d_in[4];  // (4, 1024)
  const float* pm  = (const float*)d_in[5];  // (4288, 1858)

  float* out = (float*)d_out;
  float* policy_logits = out;                  // 2048*1858
  float* pa_out = out + (size_t)BATCH * NPOL;  // 2048*64*64

  unsigned char* ws = (unsigned char*)d_ws;
  size_t off = 0;
  bf16_t* WqB = (bf16_t*)(ws + off); off += (size_t)HID * HID * 2;
  bf16_t* WkB = (bf16_t*)(ws + off); off += (size_t)HID * HID * 2;
  bf16_t* pmT = (bf16_t*)(ws + off); off += (size_t)NPOL * KPOL * 2;
  bf16_t* combined = (bf16_t*)(ws + off); off += (size_t)BATCH * KPOL * 2;
  float* Wfused = (float*)(ws + off); off += (size_t)4 * HID * 4;

  conv_w_kernel<<<(HID * HID + 255) / 256, 256, 0, stream>>>(Wq, Wk, WqB, WkB,
                                                             HID * HID);
  dim3 tgrid((NPOL + 31) / 32, (KPOL + 31) / 32);
  transpose_pm_kernel<<<tgrid, dim3(32, 8), 0, stream>>>(pm, pmT);
  wfused_kernel<<<16, 256, 0, stream>>>(Wpo, Wpk, Wfused);

  fused_attention_kernel<<<BATCH, 512, 0, stream>>>(hs, WqB, WkB, Wfused,
                                                    pa_out, combined);

  dim3 pgrid((NPOL + 127) / 128, BATCH / 128);
  policy_gemm_kernel<<<pgrid, 512, 0, stream>>>(combined, pmT, policy_logits);
}
